// MultimodalClassier_20461224198607
// MI455X (gfx1250) — compile-verified, matrix operand order checked by reading, not run on hardware
//
#include <hip/hip_runtime.h>
#include <hip/hip_bf16.h>
#include <stdint.h>

// ---------------------------------------------------------------------------
// MI455X (gfx1250, CDNA5) implementation of the VMamba/Sigma multimodal net.
// Dense math -> v_wmma_f32_16x16x32_bf16, 2x2 register-tiled, b128 frag loads,
//               WGP-scope prefetch of the streaming A operand.
// Selective scan -> templated VALU kernel, h-state in VGPRs; xdbl staged into
//               LDS in 32-step chunks (coalesced, 2 barriers per 32 steps).
// ---------------------------------------------------------------------------

typedef __attribute__((ext_vector_type(16))) __bf16 v16bf;
typedef __attribute__((ext_vector_type(8)))  __bf16 v8bf;
typedef __attribute__((ext_vector_type(8)))  float  v8f;

namespace {
constexpr int  BN = 128, DIMC = 64, HS = 16, WSZ = 16, LQ = 256, NCLS = 16;
constexpr long BL = (long)BN * LQ;                // 32768 tokens

// ---- workspace arena (byte offsets) ---------------------------------------
constexpr long SZ_XZ    = BL * 512 * 4;           // xz, up to 2*di=512 ch, f32
constexpr long SZ_NRMBF = BL * 256 * 2;           // LN output, bf16
constexpr long SZ_XCONV = BL * 256 * 4;           // conv out (b,l,di) f32 / fused / mlp hidden
constexpr long SZ_XSB   = BL * 1024 * 2;          // xs, 4 dirs, bf16 (b,k,l,d)
constexpr long SZ_XDBL  = (long)BN * 4 * 40 * LQ * 4; // x_dbl f32 (cdim<=40)
constexpr long SZ_YSB   = BL * 1024 * 2;          // ys, 4 dirs, bf16 (b,k,l,d)
constexpr long SZ_Y     = BL * 256 * 4;           // merged y f32
constexpr long SZ_GATED = BL * 256 * 2;           // gated bf16
constexpr long SZ_AH    = BL * 64 * 4;            // branch A tokens f32
constexpr long SZ_BH    = BL * 64 * 4;            // branch B tokens f32
constexpr long SZ_XZ2   = BL * 256 * 4;           // cross: second xz
constexpr long SZ_Y2    = BL * 128 * 4;           // cross: second y

constexpr long OFF_XZ    = 0;
constexpr long OFF_NRMBF = OFF_XZ    + SZ_XZ;
constexpr long OFF_XCONV = OFF_NRMBF + SZ_NRMBF;
constexpr long OFF_XSB   = OFF_XCONV + SZ_XCONV;
constexpr long OFF_XDBL  = OFF_XSB   + SZ_XSB;
constexpr long OFF_YSB   = OFF_XDBL  + SZ_XDBL;
constexpr long OFF_Y     = OFF_YSB   + SZ_YSB;
constexpr long OFF_GATED = OFF_Y     + SZ_Y;
constexpr long OFF_AH    = OFF_GATED + SZ_GATED;
constexpr long OFF_BH    = OFF_AH    + SZ_AH;
constexpr long OFF_XZ2   = OFF_BH    + SZ_BH;
constexpr long OFF_Y2    = OFF_XZ2   + SZ_XZ2;
constexpr long OFF_WBF   = OFF_Y2    + SZ_Y2;     // bf16 weight staging (~1MB)
} // namespace

// ---- bf16 helpers (RNE), no reliance on __bf16 arithmetic -----------------
__device__ __forceinline__ __bf16 f2bf(float f) {
  unsigned u = __builtin_bit_cast(unsigned, f);
  u += 0x7FFFu + ((u >> 16) & 1u);
  unsigned short h = (unsigned short)(u >> 16);
  return __builtin_bit_cast(__bf16, h);
}
__device__ __forceinline__ float bf2f(__bf16 b) {
  unsigned short h = __builtin_bit_cast(unsigned short, b);
  unsigned u = ((unsigned)h) << 16;
  return __builtin_bit_cast(float, u);
}
__device__ __forceinline__ float silu_f(float x) {
  return x / (1.f + __expf(-x));
}

// Per CDNA5 ISA 7.12.2, a lane's 16-bit 16x32 fragment is two contiguous
// 8-element chunks: k = half*8 .. +7 and k = 16+half*8 .. +7 -> two b128 loads.
__device__ __forceinline__ v16bf ldfrag(const __bf16* __restrict__ rowp, int kk, int half) {
  v8bf lo = *(const v8bf*)(rowp + kk + half * 8);
  v8bf hi = *(const v8bf*)(rowp + kk + 16 + half * 8);
  return __builtin_shufflevector(lo, hi, 0, 1, 2, 3, 4, 5, 6, 7,
                                 8, 9, 10, 11, 12, 13, 14, 15);
}

__device__ __forceinline__ void epi_tile(float* __restrict__ Cb, __bf16* __restrict__ Cx,
                                         const float* __restrict__ bias,
                                         const float* __restrict__ Rb, v8f acc,
                                         int tm, int tn, int M, int N, int ldc,
                                         int half, int lr, int act) {
#pragma unroll
  for (int r = 0; r < 8; ++r) {
    const int m = tm + r + half * 8;
    const int n = tn + lr;
    if (m < M && n < N) {
      float v = acc[r];
      if (bias) v += bias[n];
      if (act == 1) v = 0.5f * v * (1.f + erff(v * 0.70710678118f));  // exact GELU
      if (Rb) v += Rb[(long)m * ldc + n];
      Cb[(long)m * ldc + n] = v;
      if (Cx) Cx[(long)m * ldc + n] = f2bf(v);
    }
  }
}

// ---------------------------------------------------------------------------
// WMMA GEMM: C[M,N](f32) = A[M,K](bf16,lda) x B[N,K](bf16,ldb)^T
//            [+bias][gelu][+resid][bf16 mirror out]
// One wave -> 32x32 C (2x2 tiles of v_wmma_f32_16x16x32_bf16). K % 32 == 0.
// Ragged M/N handled by clamping source rows/cols + guarded stores (a C row
// depends only on its own A row, so clamping is exact). Branch-free K loop.
// Batched over gridDim.z; A offset uses (z % aMod) to reuse per-k weights.
// ---------------------------------------------------------------------------
__global__ void k_gemm_wmma(const __bf16* __restrict__ A, const __bf16* __restrict__ Bm,
                            float* __restrict__ C, __bf16* __restrict__ Cbf,
                            const float* __restrict__ bias, const float* __restrict__ resid,
                            int M, int N, int K, int lda, int ldb, int ldc,
                            int act, long sA, int aMod, long sB, long sC) {
  const int z = blockIdx.z;
  const __bf16* Ab = A + (long)(aMod ? (z % aMod) : z) * sA;
  const __bf16* Bb = Bm + (long)z * sB;
  float* Cb = C + (long)z * sC;
  __bf16* Cx = Cbf ? (Cbf + (long)z * sC) : nullptr;
  const float* Rb = resid ? (resid + (long)z * sC) : nullptr;

  const int tN = (N + 31) >> 5;
  const int tM = (M + 31) >> 5;
  const int wave = blockIdx.x * (blockDim.x >> 5) + (threadIdx.x >> 5);
  if (wave >= tM * tN) return;
  const int tm = (wave / tN) << 5;
  const int tn = (wave % tN) << 5;
  const int lane = threadIdx.x & 31;
  const int half = lane >> 4;
  const int lr = lane & 15;

  const int r0 = min(tm + lr, M - 1);
  const int r1 = min(tm + 16 + lr, M - 1);
  const int c0 = min(tn + lr, N - 1);
  const int c1 = min(tn + 16 + lr, N - 1);
  const __bf16* a0p = Ab + (long)r0 * lda;
  const __bf16* a1p = Ab + (long)r1 * lda;
  const __bf16* b0p = Bb + (long)c0 * ldb;
  const __bf16* b1p = Bb + (long)c1 * ldb;

  v8f z8 = {0.f, 0.f, 0.f, 0.f, 0.f, 0.f, 0.f, 0.f};
  v8f acc00 = z8, acc01 = z8, acc10 = z8, acc11 = z8;

  for (int kk = 0; kk < K; kk += 32) {
    const v16bf a0 = ldfrag(a0p, kk, half);
    const v16bf a1 = ldfrag(a1p, kk, half);
    const v16bf b0 = ldfrag(b0p, kk, half);
    const v16bf b1 = ldfrag(b1p, kk, half);
    // WGP-scope prefetch (locality 3) of the next k-chunk of the A stream:
    // pulled into all cache levels, consumed by this wave within the loop.
    __builtin_prefetch(a0p + kk + 32, 0, 3);
    __builtin_prefetch(a1p + kk + 32, 0, 3);
    acc00 = __builtin_amdgcn_wmma_f32_16x16x32_bf16(false, a0, false, b0, (short)0, acc00, false, false);
    acc01 = __builtin_amdgcn_wmma_f32_16x16x32_bf16(false, a0, false, b1, (short)0, acc01, false, false);
    acc10 = __builtin_amdgcn_wmma_f32_16x16x32_bf16(false, a1, false, b0, (short)0, acc10, false, false);
    acc11 = __builtin_amdgcn_wmma_f32_16x16x32_bf16(false, a1, false, b1, (short)0, acc11, false, false);
  }
  epi_tile(Cb, Cx, bias, Rb, acc00, tm,      tn,      M, N, ldc, half, lr, act);
  epi_tile(Cb, Cx, bias, Rb, acc01, tm,      tn + 16, M, N, ldc, half, lr, act);
  epi_tile(Cb, Cx, bias, Rb, acc10, tm + 16, tn,      M, N, ldc, half, lr, act);
  epi_tile(Cb, Cx, bias, Rb, acc11, tm + 16, tn + 16, M, N, ldc, half, lr, act);
}

// ---- f32 -> bf16 conversion (weight staging) ------------------------------
__global__ void k_cvt_bf16(const float* __restrict__ in, __bf16* __restrict__ out, long n) {
  long i = (long)blockIdx.x * blockDim.x + threadIdx.x;
  if (i < n) out[i] = f2bf(in[i]);
}

// ---- LayerNorm over last dim; optional f32 + bf16 output, channel offset ---
__global__ void k_ln(const float* __restrict__ x, const float* __restrict__ g,
                     const float* __restrict__ b, float* __restrict__ outF,
                     __bf16* __restrict__ outB, int C, int outLd, int outOff) {
  const int row = blockIdx.x;
  const float* xr = x + (long)row * C;
  __shared__ float s1[128], s2[128];
  float sum = 0.f, sq = 0.f;
  for (int c = threadIdx.x; c < C; c += blockDim.x) {
    float v = xr[c];
    sum += v; sq += v * v;
  }
  s1[threadIdx.x] = sum; s2[threadIdx.x] = sq;
  __syncthreads();
  for (int st = blockDim.x >> 1; st > 0; st >>= 1) {
    if ((int)threadIdx.x < st) { s1[threadIdx.x] += s1[threadIdx.x + st];
                                 s2[threadIdx.x] += s2[threadIdx.x + st]; }
    __syncthreads();
  }
  const float mean = s1[0] / C;
  const float inv = rsqrtf(s2[0] / C - mean * mean + 1e-5f);
  for (int c = threadIdx.x; c < C; c += blockDim.x) {
    float v = (xr[c] - mean) * inv * g[c] + b[c];
    long o = (long)row * outLd + outOff + c;
    if (outF) outF[o] = v;
    if (outB) outB[o] = f2bf(v);
  }
}

// ---- depthwise 3x3 conv ('SAME') + bias + SiLU; xz (BL,2di) -> (b,l,di) ----
__global__ void k_dwconv_silu(const float* __restrict__ xz, const float* __restrict__ w,
                              const float* __restrict__ bias, float* __restrict__ xconv,
                              int di) {
  long idx = (long)blockIdx.x * blockDim.x + threadIdx.x;
  long tot = (long)BN * di * LQ;
  if (idx >= tot) return;
  int c = idx % di; long t = idx / di;
  int wq = t % WSZ; t /= WSZ;
  int hq = t % HS;
  int b = (int)(t / HS);
  const int twodi = 2 * di;
  float s = bias[c];
#pragma unroll
  for (int kh = 0; kh < 3; ++kh) {
    int hh = hq + kh - 1; if (hh < 0 || hh >= HS) continue;
#pragma unroll
    for (int kw = 0; kw < 3; ++kw) {
      int ww = wq + kw - 1; if (ww < 0 || ww >= WSZ) continue;
      s += xz[((long)b * LQ + hh * WSZ + ww) * twodi + c] * w[c * 9 + kh * 3 + kw];
    }
  }
  xconv[((long)b * LQ + hq * WSZ + wq) * di + c] = s * (1.f / (1.f + __expf(-s)));
}

// ---- build 4 scan directions (bf16), token-major layout (b,k,l,d) ----------
__global__ void k_build_xs(const float* __restrict__ xconv, __bf16* __restrict__ xs, int di) {
  long idx = (long)blockIdx.x * blockDim.x + threadIdx.x;
  long tot = (long)BN * LQ * di;
  if (idx >= tot) return;
  int d = idx % di; long t = idx / di;
  int l = t % LQ; int b = (int)(t / LQ);
  int hq = l / WSZ, wq = l % WSZ;
  int l2 = LQ - 1 - l; int h2 = l2 / WSZ, w2 = l2 % WSZ;
  const float* src = xconv + (long)b * LQ * di;                 // (l, d)
  long base = ((long)(b * 4 + 0) * LQ + l) * di + d;
  long stK = (long)LQ * di;
  xs[base]           = f2bf(src[(long)l * di + d]);             // h-w order
  xs[base + stK]     = f2bf(src[(long)(wq * HS + hq) * di + d]);// w-h order (H==W)
  xs[base + 2 * stK] = f2bf(src[(long)(LQ - 1 - l) * di + d]);  // reversed h-w
  xs[base + 3 * stK] = f2bf(src[(long)(w2 * HS + h2) * di + d]);// reversed w-h
}

// ---- selective scan: one block per (b,k), one thread per d -----------------
// delta (softplus of dt proj) fused; state h[NS] fully in VGPRs (templated).
// xdbl is staged into LDS in 32-step chunks with coalesced row-major loads:
// 2 barriers per 32 steps instead of 2 per step; compute reads are LDS
// broadcasts (all lanes read the same word -> conflict-free).
template <int NS, int DTR>
__global__ void k_scan(const __bf16* __restrict__ xs, const float* __restrict__ xdbl,
                       const float* __restrict__ A_log, const float* __restrict__ dt_w,
                       const float* __restrict__ dt_b, const float* __restrict__ Ds,
                       __bf16* __restrict__ ys, int di) {
  constexpr int CD = DTR + 2 * NS;    // rows of x_dbl (dts | B | C)
  constexpr int CH = 32;              // time-steps per LDS chunk
  const int bk = blockIdx.x;          // b*4 + k
  const int k = bk & 3;
  const int d = threadIdx.x;
  __shared__ float sX[CD][CH];

  float Areg[NS], dtw[DTR], h[NS];
#pragma unroll
  for (int i = 0; i < NS; ++i) {
    Areg[i] = -__expf(A_log[((long)k * di + d) * NS + i]);
    h[i] = 0.f;
  }
#pragma unroll
  for (int r = 0; r < DTR; ++r) dtw[r] = dt_w[((long)k * di + d) * DTR + r];
  const float dtb = dt_b[(long)k * di + d];
  const float Dk = Ds[(long)k * di + d];
  const float* xrow = xdbl + (long)bk * CD * LQ;

  for (int l0 = 0; l0 < LQ; l0 += CH) {
    // stage (CD x CH) tile: contiguous in l within each row -> coalesced
    for (int t = threadIdx.x; t < CD * CH; t += blockDim.x) {
      const int c = t / CH, j = t % CH;
      sX[c][j] = xrow[(long)c * LQ + l0 + j];
    }
    __syncthreads();
#pragma unroll 4
    for (int j = 0; j < CH; ++j) {
      const int l = l0 + j;
      const long tok = ((long)bk * LQ + l) * di + d;   // token-major, coalesced
      const float u = bf2f(xs[tok]);
      float sdt = dtb;
#pragma unroll
      for (int r = 0; r < DTR; ++r) sdt += dtw[r] * sX[r][j];
      const float dl = (sdt > 20.f) ? sdt : log1pf(__expf(sdt));   // softplus
      const float du = dl * u;
      float y = 0.f;
#pragma unroll
      for (int i = 0; i < NS; ++i) {
        const float dA = __expf(dl * Areg[i]);
        h[i] = h[i] * dA + du * sX[DTR + i][j];
        y += h[i] * sX[DTR + NS + i][j];
      }
      ys[tok] = f2bf(y + u * Dk);
    }
    __syncthreads();
  }
}

// ---- merge 4 directions -> (BL, di) f32 ------------------------------------
__global__ void k_merge_dirs(const __bf16* __restrict__ ys, float* __restrict__ y, int di) {
  long idx = (long)blockIdx.x * blockDim.x + threadIdx.x;
  long tot = BL * di;
  if (idx >= tot) return;
  int d = idx % di; long t = idx / di;
  int l = t % LQ; int b = (int)(t / LQ);
  int hq = l / WSZ, wq = l % WSZ;
  int lw = wq * HS + hq;
  long b4 = (long)b * 4;
  long i0 = ((b4 + 0) * LQ + l) * di + d;
  long i1 = ((b4 + 1) * LQ + lw) * di + d;
  long i2 = ((b4 + 2) * LQ + (LQ - 1 - l)) * di + d;
  long i3 = ((b4 + 3) * LQ + (LQ - 1 - lw)) * di + d;
  y[idx] = bf2f(ys[i0]) + bf2f(ys[i2]) + bf2f(ys[i1]) + bf2f(ys[i3]);
}

// ---- gating: g = y * silu(z), bf16 out -------------------------------------
__global__ void k_gate(const float* __restrict__ y, const float* __restrict__ xz,
                       __bf16* __restrict__ g, int di, int zld, int zoff, long tot) {
  long idx = (long)blockIdx.x * blockDim.x + threadIdx.x;
  if (idx >= tot) return;
  int d = idx % di; long row = idx / di;
  g[idx] = f2bf(y[idx] * silu_f(xz[row * zld + zoff + d]));
}

__global__ void k_add(float* __restrict__ dst, const float* __restrict__ src, long n) {
  long i = (long)blockIdx.x * blockDim.x + threadIdx.x;
  if (i < n) dst[i] += src[i];
}

// ---- x1 (B,C,H,W) -> tokens (BL,C) -----------------------------------------
__global__ void k_nchw_tokens(const float* __restrict__ x, float* __restrict__ out) {
  long idx = (long)blockIdx.x * blockDim.x + threadIdx.x;
  long tot = BL * DIMC;
  if (idx >= tot) return;
  int c = idx % DIMC; long t = idx / DIMC;
  int l = t % LQ; int b = (int)(t / LQ);
  out[idx] = x[((long)b * DIMC + c) * LQ + l];
}

// ---- stem: 1x1 conv + BN + ReLU on x2 (B,1,H,W) -> tokens (BL,64) ----------
__global__ void k_stem(const float* __restrict__ x2, const float* __restrict__ w,
                       const float* __restrict__ bconv, const float* __restrict__ g,
                       const float* __restrict__ beta, const float* __restrict__ mean,
                       const float* __restrict__ var, float* __restrict__ out) {
  long idx = (long)blockIdx.x * blockDim.x + threadIdx.x;
  long tot = BL * DIMC;
  if (idx >= tot) return;
  int c = idx % DIMC; long t = idx / DIMC;
  int l = t % LQ; int b = (int)(t / LQ);
  float v = x2[(long)b * LQ + l] * w[c] + bconv[c];
  float sc = g[c] * rsqrtf(var[c] + 1e-3f);
  float r = v * sc + beta[c] - mean[c] * sc;
  out[idx] = fmaxf(r, 0.f);
}

// ---- mean pool over L + head GEMV ------------------------------------------
__global__ void k_head(const float* __restrict__ fused, const float* __restrict__ hw,
                       float* __restrict__ out) {
  const int b = blockIdx.x;
  __shared__ float pooled[DIMC];
  float s = 0.f;
  for (int l = 0; l < LQ; ++l) s += fused[((long)b * LQ + l) * DIMC + threadIdx.x];
  pooled[threadIdx.x] = s * (1.f / LQ);
  __syncthreads();
  if ((int)threadIdx.x < NCLS) {
    float o = 0.f;
    for (int c = 0; c < DIMC; ++c) o += pooled[c] * hw[threadIdx.x * DIMC + c];
    out[(long)b * NCLS + threadIdx.x] = o;
  }
}

// ===========================================================================
// Host orchestration
// ===========================================================================
namespace {

dim3 gblk(long tot) { return dim3((unsigned)((tot + 255) / 256)); }

void gemm(hipStream_t s, const __bf16* A, const __bf16* Bm, float* C, __bf16* Cbf,
          const float* bias, const float* resid, int M, int N, int K,
          int lda, int ldb, int ldc, int act,
          int batch = 1, long sA = 0, int aMod = 0, long sB = 0, long sC = 0) {
  int tiles = ((M + 31) / 32) * ((N + 31) / 32);
  int blocks = (tiles + 3) / 4;                       // 4 waves per block
  k_gemm_wmma<<<dim3(blocks, 1, batch), dim3(128), 0, s>>>(
      A, Bm, C, Cbf, bias, resid, M, N, K, lda, ldb, ldc, act, sA, aMod, sB, sC);
}

const __bf16* toBF(const float* src, long n, char*& cur, hipStream_t s) {
  uintptr_t p = ((uintptr_t)cur + 15) & ~(uintptr_t)15;
  __bf16* dst = (__bf16*)p;
  cur = (char*)(p + (size_t)n * 2);
  k_cvt_bf16<<<gblk(n), dim3(256), 0, s>>>(src, dst, n);
  return dst;
}

struct SS2DP {
  const __bf16 *inp, *xproj, *outp;
  const float *convw, *convb, *dtw, *dtb, *alog, *ds, *outg, *outb;
  int dim, di, dtr, ns, cdim, outDim;
};

SS2DP mkSS2D(void* const* din, int base, int dim, int ns, int outDim,
             char*& cur, hipStream_t s) {
  SS2DP p;
  p.dim = dim; p.di = 2 * dim; p.dtr = (dim + 15) / 16;
  p.ns = ns; p.cdim = p.dtr + 2 * ns; p.outDim = outDim;
  p.convw = (const float*)din[base + 1];
  p.convb = (const float*)din[base + 2];
  p.dtw   = (const float*)din[base + 4];
  p.dtb   = (const float*)din[base + 5];
  p.alog  = (const float*)din[base + 6];
  p.ds    = (const float*)din[base + 7];
  p.outg  = (const float*)din[base + 8];
  p.outb  = (const float*)din[base + 9];
  p.inp   = toBF((const float*)din[base + 0], (long)2 * p.di * dim, cur, s);
  p.xproj = toBF((const float*)din[base + 3], (long)4 * p.cdim * p.di, cur, s);
  p.outp  = toBF((const float*)din[base + 10], (long)outDim * p.di, cur, s);
  return p;
}

struct VSSP {
  SS2DP ss;
  const float *ln1g, *ln1b, *ln2g, *ln2b, *b1, *b2;
  const __bf16 *w1, *w2;
};

VSSP mkVSS(void* const* din, int base, char*& cur, hipStream_t s) {
  VSSP v;
  v.ln1g = (const float*)din[base + 0];
  v.ln1b = (const float*)din[base + 1];
  v.ss   = mkSS2D(din, base + 2, DIMC, 16, DIMC, cur, s);
  v.ln2g = (const float*)din[base + 13];
  v.ln2b = (const float*)din[base + 14];
  v.w1   = toBF((const float*)din[base + 15], (long)2 * DIMC * DIMC, cur, s);
  v.b1   = (const float*)din[base + 16];
  v.w2   = toBF((const float*)din[base + 17], (long)DIMC * 2 * DIMC, cur, s);
  v.b2   = (const float*)din[base + 18];
  return v;
}

// pre-gate SS2D core: nrm(BL,dim) bf16 -> xz (BL,2di) f32, y (BL,di) f32 (LN'd)
void ss2d_core(hipStream_t s, const SS2DP& p, const __bf16* nrm, float* xz,
               float* y, char* ws) {
  float*  xconv = (float*)(ws + OFF_XCONV);
  __bf16* xsb   = (__bf16*)(ws + OFF_XSB);
  float*  xdbl  = (float*)(ws + OFF_XDBL);
  __bf16* ysb   = (__bf16*)(ws + OFF_YSB);
  const int di = p.di, cdim = p.cdim;

  // xz = nrm @ in_proj^T  (B = in_proj (2di, dim), K-contiguous)
  gemm(s, nrm, p.inp, xz, nullptr, nullptr, nullptr,
       (int)BL, 2 * di, p.dim, p.dim, p.dim, 2 * di, 0);
  // depthwise conv + SiLU -> (b,l,di)
  k_dwconv_silu<<<gblk((long)BN * di * LQ), dim3(256), 0, s>>>(xz, p.convw, p.convb, xconv, di);
  // 4 scan directions (bf16, token-major)
  k_build_xs<<<gblk((long)BN * LQ * di), dim3(256), 0, s>>>(xconv, xsb, di);
  // x_dbl[bk](cdim,L) = x_proj[k](cdim,di) x xs[bk](L,di)^T; batched, A mod 4
  gemm(s, p.xproj, xsb, xdbl, nullptr, nullptr, nullptr,
       cdim, LQ, di, di, di, LQ, 0,
       BN * 4, (long)cdim * di, 4, (long)LQ * di, (long)cdim * LQ);
  // selective scan (delta fused)
  dim3 g(BN * 4), b(di);
  if (p.ns == 16 && p.dtr == 4)
    k_scan<16, 4><<<g, b, 0, s>>>(xsb, xdbl, p.alog, p.dtw, p.dtb, p.ds, ysb, di);
  else if (p.ns == 4 && p.dtr == 4)
    k_scan<4, 4><<<g, b, 0, s>>>(xsb, xdbl, p.alog, p.dtw, p.dtb, p.ds, ysb, di);
  else
    k_scan<16, 8><<<g, b, 0, s>>>(xsb, xdbl, p.alog, p.dtw, p.dtb, p.ds, ysb, di);
  // merge directions -> (BL,di), then LN in place
  k_merge_dirs<<<gblk(BL * di), dim3(256), 0, s>>>(ysb, y, di);
  k_ln<<<dim3((unsigned)BL), dim3(128), 0, s>>>(y, p.outg, p.outb, y, nullptr, di, di, 0);
}

// full SS2D: core + self-gate + out_proj (+ residual)
void ss2d_full(hipStream_t s, const SS2DP& p, const __bf16* nrm, float* xz, float* y,
               const float* resid, float* dst, char* ws) {
  ss2d_core(s, p, nrm, xz, y, ws);
  __bf16* gated = (__bf16*)(ws + OFF_GATED);
  k_gate<<<gblk(BL * p.di), dim3(256), 0, s>>>(y, xz, gated, p.di, 2 * p.di, p.di, BL * p.di);
  gemm(s, gated, p.outp, dst, nullptr, nullptr, resid,
       (int)BL, p.outDim, p.di, p.di, p.di, p.outDim, 0);
}

void vss_run(hipStream_t s, const VSSP& v, float* xh, char* ws) {
  __bf16* nrm = (__bf16*)(ws + OFF_NRMBF);
  float* xz = (float*)(ws + OFF_XZ);
  float* y = (float*)(ws + OFF_Y);
  k_ln<<<dim3((unsigned)BL), dim3(128), 0, s>>>(xh, v.ln1g, v.ln1b, nullptr, nrm, DIMC, DIMC, 0);
  ss2d_full(s, v.ss, nrm, xz, y, xh, xh, ws);
  k_ln<<<dim3((unsigned)BL), dim3(128), 0, s>>>(xh, v.ln2g, v.ln2b, nullptr, nrm, DIMC, DIMC, 0);
  float*  m1f = (float*)(ws + OFF_XCONV);
  __bf16* m1b = (__bf16*)(ws + OFF_GATED);
  gemm(s, nrm, v.w1, m1f, m1b, v.b1, nullptr, (int)BL, 2 * DIMC, DIMC, DIMC, DIMC, 2 * DIMC, 1);
  gemm(s, m1b, v.w2, xh, nullptr, v.b2, xh, (int)BL, DIMC, 2 * DIMC, 2 * DIMC, 2 * DIMC, DIMC, 0);
}

} // namespace

extern "C" void kernel_launch(void* const* d_in, const int* in_sizes, int n_in,
                              void* d_out, int out_size, void* d_ws, size_t ws_size,
                              hipStream_t stream) {
  (void)in_sizes; (void)n_in; (void)out_size; (void)ws_size;
  char* ws = (char*)d_ws;
  hipStream_t s = stream;

  // ---- stage bf16 weights --------------------------------------------------
  char* cur = ws + OFF_WBF;
  VSSP vss[4];
  const int vbase[4] = {8, 27, 46, 65};
  for (int i = 0; i < 4; ++i) vss[i] = mkVSS(d_in, vbase[i], cur, s);
  SS2DP ssa = mkSS2D(d_in, 88, DIMC, 4, DIMC, cur, s);
  SS2DP ssb = mkSS2D(d_in, 99, DIMC, 4, DIMC, cur, s);
  SS2DP ssc = mkSS2D(d_in, 114, 2 * DIMC, 16, DIMC, cur, s);

  float* AH = (float*)(ws + OFF_AH);
  float* BH = (float*)(ws + OFF_BH);

  // ---- stems ---------------------------------------------------------------
  k_nchw_tokens<<<gblk(BL * DIMC), dim3(256), 0, s>>>((const float*)d_in[0], AH);
  k_stem<<<gblk(BL * DIMC), dim3(256), 0, s>>>(
      (const float*)d_in[1], (const float*)d_in[2], (const float*)d_in[3],
      (const float*)d_in[4], (const float*)d_in[5], (const float*)d_in[6],
      (const float*)d_in[7], BH);

  // ---- 4 shared VSS blocks on both branches --------------------------------
  for (int i = 0; i < 4; ++i) {
    vss_run(s, vss[i], AH, ws);
    vss_run(s, vss[i], BH, ws);
  }

  // ---- cross-modal SS2D (swapped gates) ------------------------------------
  {
    __bf16* nrm = (__bf16*)(ws + OFF_NRMBF);
    float* xzA = (float*)(ws + OFF_XZ);
    float* xzB = (float*)(ws + OFF_XZ2);
    float* yA = (float*)(ws + OFF_Y);
    float* yB = (float*)(ws + OFF_Y2);
    __bf16* gated = (__bf16*)(ws + OFF_GATED);
    k_ln<<<dim3((unsigned)BL), dim3(128), 0, s>>>(AH, (const float*)d_in[84],
        (const float*)d_in[85], nullptr, nrm, DIMC, DIMC, 0);
    ss2d_core(s, ssa, nrm, xzA, yA, ws);
    k_ln<<<dim3((unsigned)BL), dim3(128), 0, s>>>(BH, (const float*)d_in[86],
        (const float*)d_in[87], nullptr, nrm, DIMC, DIMC, 0);
    ss2d_core(s, ssb, nrm, xzB, yB, ws);
    // oa = (ya * silu(zb)) @ out_proj_a^T + AH
    k_gate<<<gblk(BL * 128), dim3(256), 0, s>>>(yA, xzB, gated, 128, 256, 128, BL * 128);
    gemm(s, gated, ssa.outp, AH, nullptr, nullptr, AH, (int)BL, DIMC, 128, 128, 128, DIMC, 0);
    // ob = (yb * silu(za)) @ out_proj_b^T + BH
    k_gate<<<gblk(BL * 128), dim3(256), 0, s>>>(yB, xzA, gated, 128, 256, 128, BL * 128);
    gemm(s, gated, ssb.outp, BH, nullptr, nullptr, BH, (int)BL, DIMC, 128, 128, 128, DIMC, 0);
  }

  // ---- concat fusion SS2D (dim=128) ----------------------------------------
  float* fused = (float*)(ws + OFF_XCONV);  // written at the very end of ss2d_full
  {
    __bf16* nrm = (__bf16*)(ws + OFF_NRMBF);
    float* xz = (float*)(ws + OFF_XZ);
    float* y = (float*)(ws + OFF_Y);
    k_ln<<<dim3((unsigned)BL), dim3(128), 0, s>>>(AH, (const float*)d_in[110],
        (const float*)d_in[111], nullptr, nrm, DIMC, 128, 0);
    k_ln<<<dim3((unsigned)BL), dim3(128), 0, s>>>(BH, (const float*)d_in[112],
        (const float*)d_in[113], nullptr, nrm, DIMC, 128, DIMC);
    ss2d_full(s, ssc, nrm, xz, y, AH, fused, ws);     // fused = y_out + AH
    k_add<<<gblk(BL * DIMC), dim3(256), 0, s>>>(fused, BH, BL * DIMC);  // + BH
  }

  // ---- pool + head ---------------------------------------------------------
  k_head<<<dim3(BN), dim3(DIMC), 0, s>>>(fused, (const float*)d_in[125], (float*)d_out);
}
